// MambaBlock_49950469652533
// MI455X (gfx1250) — matrix operand order repaired and checked by reading, not compiled, not run on hardware
//
#include <hip/hip_runtime.h>

// ---- problem constants (match reference) ----
#define D_MODEL 1024
#define D_STATE 16
#define D_CONV  4
#define D_INNER 2048           // EXPAND * D_MODEL
#define DT_RANK 64
#define BB 2
#define LL 2048
#define MM (BB * LL)           // 4096 rows for all GEMMs

typedef __attribute__((ext_vector_type(16))) __bf16 bf16x16;
typedef __attribute__((ext_vector_type(8)))  float  floatx8;
typedef __attribute__((ext_vector_type(4)))  int    v4i;

struct U8 { uint4 a; uint4 b; };   // 32 bytes == bf16x16

__device__ __forceinline__ unsigned short f2bf(float f) {
  unsigned int u = __builtin_bit_cast(unsigned int, f);
  u += 0x7FFFu + ((u >> 16) & 1u);               // round-to-nearest-even
  return (unsigned short)(u >> 16);
}

// ---------------- CDNA5 async LDS copy (guarded) ----------------
#if __has_builtin(__builtin_amdgcn_global_load_async_to_lds_b128)
#define HAS_ASYNC_LDS 1
#else
#define HAS_ASYNC_LDS 0
#endif

typedef v4i __attribute__((address_space(1)))* gptr128;
typedef v4i __attribute__((address_space(3)))* lptr128;

__device__ __forceinline__ void async_copy_b128(const unsigned short* g, unsigned short* l) {
#if HAS_ASYNC_LDS
  __builtin_amdgcn_global_load_async_to_lds_b128((gptr128)g, (lptr128)l, 0, 0);
#else
  *(uint4*)l = *(const uint4*)g;
#endif
}

__device__ __forceinline__ void wait_async() {
#if HAS_ASYNC_LDS
#if __has_builtin(__builtin_amdgcn_s_wait_asynccnt)
  __builtin_amdgcn_s_wait_asynccnt(0);
#else
  asm volatile("s_wait_asynccnt 0x0" ::: "memory");
#endif
#endif
}

// =====================================================================
// GEMM: C[M,N] = A[M,K] * W[N,K]^T   (A, W bf16 row-major; C fp32)
// Block tile 128x128, BK=64, 8 waves (4x2), wave tile 32x64.
// Double-buffered LDS, async global->LDS pipeline.
// =====================================================================
#define BM 128
#define BN 128
#define BK 64

__global__ __launch_bounds__(256)
void gemm_bf16_wmma(const unsigned short* __restrict__ A, int lda,
                    const unsigned short* __restrict__ W, int ldw,
                    float* __restrict__ C, int ldc,
                    int N, int K)
{
  __shared__ unsigned short sA[2][BM][BK];   // 32 KB
  __shared__ unsigned short sB[2][BN][BK];   // 32 KB

  const int tid  = threadIdx.x;
  const int wave = tid >> 5;
  const int lane = tid & 31;
  const int h    = lane >> 4;          // half-wave select
  const int ln   = lane & 15;
  const int wm   = (wave >> 1) << 5;   // 0,32,64,96
  const int wn   = (wave & 1) << 6;    // 0,64
  const int m0   = blockIdx.x * BM;
  const int n0   = blockIdx.y * BN;

  // staging: thread -> (row 0..127, 32 cols = 64B)
  const int srow = tid >> 1;
  const int scol = (tid & 1) * 32;
  const int gn   = n0 + srow;

  floatx8 acc[2][4] = {};

  auto stage = [&](int buf, int k0) {
    const unsigned short* ga = A + (size_t)(m0 + srow) * lda + k0 + scol;
    unsigned short* la = &sA[buf][srow][scol];
#pragma unroll
    for (int i = 0; i < 4; ++i) async_copy_b128(ga + i * 8, la + i * 8);
    unsigned short* lb = &sB[buf][srow][scol];
    if (gn < N) {
      const unsigned short* gb = W + (size_t)gn * ldw + k0 + scol;
#pragma unroll
      for (int i = 0; i < 4; ++i) async_copy_b128(gb + i * 8, lb + i * 8);
    } else {
      const uint4 z = {0u, 0u, 0u, 0u};
#pragma unroll
      for (int i = 0; i < 4; ++i) *(uint4*)(lb + i * 8) = z;
    }
  };

  stage(0, 0);
  int buf = 0;
  for (int k0 = 0; k0 < K; k0 += BK) {
    wait_async();
    __syncthreads();                      // staged tiles visible to all
    if (k0 + BK < K) stage(buf ^ 1, k0 + BK);

#pragma unroll
    for (int kc = 0; kc < 2; ++kc) {
      // A fragments (16-bit A layout: uints {4h..} and {8+4h..} of 32-K chunk)
      bf16x16 afr[2];
#pragma unroll
      for (int sm = 0; sm < 2; ++sm) {
        const unsigned int* rp = (const unsigned int*)&sA[buf][wm + sm * 16 + ln][kc * 32];
        U8 t;
        t.a = *(const uint4*)(rp + 4 * h);
        t.b = *(const uint4*)(rp + 8 + 4 * h);
        afr[sm] = __builtin_bit_cast(bf16x16, t);
      }
#pragma unroll
      for (int sn = 0; sn < 4; ++sn) {
        // B fragment (same K order as A: uints {4h..} and {8+4h..} of 32-K chunk)
        const unsigned int* rp = (const unsigned int*)&sB[buf][wn + sn * 16 + ln][kc * 32];
        U8 t;
        t.a = *(const uint4*)(rp + 4 * h);
        t.b = *(const uint4*)(rp + 8 + 4 * h);
        bf16x16 bfr = __builtin_bit_cast(bf16x16, t);
        acc[0][sn] = __builtin_amdgcn_wmma_f32_16x16x32_bf16(
            false, afr[0], false, bfr, (short)0, acc[0][sn], false, false);
        acc[1][sn] = __builtin_amdgcn_wmma_f32_16x16x32_bf16(
            false, afr[1], false, bfr, (short)0, acc[1][sn], false, false);
      }
    }
    buf ^= 1;
  }

  // store: VGPR r -> (M = r + 8h, N = ln)
#pragma unroll
  for (int sm = 0; sm < 2; ++sm) {
#pragma unroll
    for (int sn = 0; sn < 4; ++sn) {
      const int n = n0 + wn + sn * 16 + ln;
      if (n < N) {
#pragma unroll
        for (int r = 0; r < 8; ++r) {
          const int m = m0 + wm + sm * 16 + r + 8 * h;
          C[(size_t)m * ldc + n] = acc[sm][sn][r];
        }
      }
    }
  }
}

// =====================================================================
// fp32 -> bf16 bulk conversion (8 elements / thread, packed stores)
// =====================================================================
__global__ __launch_bounds__(256)
void cvt_bf16_kernel(const float* __restrict__ in, unsigned short* __restrict__ out, int n8)
{
  const int i = blockIdx.x * 256 + threadIdx.x;
  if (i >= n8) return;
  const float4* p = (const float4*)in + (size_t)i * 2;
  const float4 a = p[0], b = p[1];
  uint4 r;
  r.x = (unsigned)f2bf(a.x) | ((unsigned)f2bf(a.y) << 16);
  r.y = (unsigned)f2bf(a.z) | ((unsigned)f2bf(a.w) << 16);
  r.z = (unsigned)f2bf(b.x) | ((unsigned)f2bf(b.y) << 16);
  r.w = (unsigned)f2bf(b.z) | ((unsigned)f2bf(b.w) << 16);
  ((uint4*)out)[i] = r;
}

// =====================================================================
// Causal depthwise conv1d (w=4) + bias + SiLU -> u (fp32 for scan, bf16 for GEMM)
// =====================================================================
__global__ __launch_bounds__(256)
void conv_silu_kernel(const float* __restrict__ xz,
                      const float* __restrict__ conv_w,
                      const float* __restrict__ conv_b,
                      float* __restrict__ u,
                      unsigned short* __restrict__ ubf)
{
  const int idx = blockIdx.x * 256 + threadIdx.x;   // over MM * D_INNER
  const int d  = idx & (D_INNER - 1);
  const int bl = idx >> 11;                          // D_INNER = 2^11
  const int t  = bl & (LL - 1);
  float acc = conv_b[d];
#pragma unroll
  for (int j = 0; j < D_CONV; ++j) {
    const int tt = t - (D_CONV - 1) + j;
    if (tt >= 0)
      acc += conv_w[d * D_CONV + j] *
             xz[(size_t)(bl - (D_CONV - 1) + j) * (2 * D_INNER) + d];
  }
  const float v = acc / (1.0f + __expf(-acc));       // SiLU
  u[idx]   = v;
  ubf[idx] = f2bf(v);
}

// =====================================================================
// Selective scan. One thread per channel d; h[16] in registers; B_t/C_t
// broadcast via double-buffered LDS (one barrier per timestep).
// Fuses: softplus(dt_raw + bias), y = (scan + u*D) * silu(z) -> bf16.
// =====================================================================
__global__ __launch_bounds__(256)
void scan_kernel(const float* __restrict__ u,
                 const float* __restrict__ dtraw,
                 const float* __restrict__ dt_bias,
                 const float* __restrict__ xdbl,
                 const float* __restrict__ A_log,
                 const float* __restrict__ Dp,
                 const float* __restrict__ xz,
                 unsigned short* __restrict__ ybf)
{
  const int b = blockIdx.y;
  const int d = blockIdx.x * 256 + threadIdx.x;

  float An[D_STATE];
#pragma unroll
  for (int n = 0; n < D_STATE; ++n) An[n] = -__expf(A_log[d * D_STATE + n]);
  float hS[D_STATE];
#pragma unroll
  for (int n = 0; n < D_STATE; ++n) hS[n] = 0.f;

  const float bias = dt_bias[d];
  const float Dd   = Dp[d];

  __shared__ float sBC[2][2 * D_STATE];

  for (int t = 0; t < LL; ++t) {
    const int p = t & 1;
    const size_t row = (size_t)(b * LL + t);
    if (threadIdx.x < 2 * D_STATE)
      sBC[p][threadIdx.x] = xdbl[row * 96 + DT_RANK + threadIdx.x];
    __syncthreads();

    const float ut = u[row * D_INNER + d];
    const float dr = dtraw[row * D_INNER + d] + bias;
    const float dt = (dr > 20.f) ? dr : log1pf(__expf(dr));   // softplus
    const float du = dt * ut;
    float yt = 0.f;
#pragma unroll
    for (int n = 0; n < D_STATE; ++n) {
      const float dA = __expf(dt * An[n]);
      hS[n] = dA * hS[n] + du * sBC[p][n];
      yt += hS[n] * sBC[p][D_STATE + n];
    }
    const float zt = xz[row * (2 * D_INNER) + D_INNER + d];
    const float yv = (yt + ut * Dd) * (zt / (1.f + __expf(-zt)));
    ybf[row * D_INNER + d] = f2bf(yv);
  }
}

// =====================================================================
extern "C" void kernel_launch(void* const* d_in, const int* in_sizes, int n_in,
                              void* d_out, int out_size, void* d_ws, size_t ws_size,
                              hipStream_t stream)
{
  (void)in_sizes; (void)n_in; (void)out_size; (void)ws_size;
  const float* x        = (const float*)d_in[0];
  const float* in_proj  = (const float*)d_in[1];
  const float* conv_w   = (const float*)d_in[2];
  const float* conv_b   = (const float*)d_in[3];
  const float* x_proj   = (const float*)d_in[4];
  const float* dt_proj  = (const float*)d_in[5];
  const float* dt_bias  = (const float*)d_in[6];
  const float* A_log    = (const float*)d_in[7];
  const float* Dp       = (const float*)d_in[8];
  const float* out_proj = (const float*)d_in[9];
  float* out = (float*)d_out;

  // ---- workspace layout ----
  float* ws    = (float*)d_ws;
  float* xz    = ws;                                   // MM * 4096 fp32
  float* uf    = xz    + (size_t)MM * 2 * D_INNER;     // MM * 2048 fp32
  float* xdbl  = uf    + (size_t)MM * D_INNER;         // MM * 96 fp32
  float* dtraw = xdbl  + (size_t)MM * 96;              // MM * 2048 fp32
  unsigned short* hb = (unsigned short*)(dtraw + (size_t)MM * D_INNER);
  unsigned short* xbf    = hb;                                   // MM*1024
  unsigned short* ubf    = xbf    + (size_t)MM * D_MODEL;        // MM*2048
  unsigned short* xdblbf = ubf    + (size_t)MM * D_INNER;        // MM*96
  unsigned short* ybf    = xdblbf + (size_t)MM * 96;             // MM*2048
  unsigned short* w_in   = ybf    + (size_t)MM * D_INNER;        // 4096*1024
  unsigned short* w_xp   = w_in   + (size_t)2 * D_INNER * D_MODEL;          // 96*2048
  unsigned short* w_dt   = w_xp   + (size_t)(DT_RANK + 2 * D_STATE) * D_INNER; // 2048*64
  unsigned short* w_out  = w_dt   + (size_t)D_INNER * DT_RANK;   // 1024*2048

  dim3 blk(256);
  auto cvt = [&](const float* src, unsigned short* dst, size_t n) {
    const int n8 = (int)(n / 8);
    cvt_bf16_kernel<<<dim3((n8 + 255) / 256), blk, 0, stream>>>(src, dst, n8);
  };

  // 0) one-time bf16 conversions (operands reused by whole GEMM tiles)
  cvt(x,        xbf,  (size_t)MM * D_MODEL);
  cvt(in_proj,  w_in, (size_t)2 * D_INNER * D_MODEL);
  cvt(x_proj,   w_xp, (size_t)(DT_RANK + 2 * D_STATE) * D_INNER);
  cvt(dt_proj,  w_dt, (size_t)D_INNER * DT_RANK);
  cvt(out_proj, w_out,(size_t)D_MODEL * D_INNER);

  // 1) in_proj: xz = x @ in_proj_w^T   (M=4096, N=4096, K=1024)
  gemm_bf16_wmma<<<dim3(MM / BM, (2 * D_INNER) / BN), blk, 0, stream>>>(
      xbf, D_MODEL, w_in, D_MODEL, xz, 2 * D_INNER, 2 * D_INNER, D_MODEL);

  // 2) conv + SiLU -> u (fp32 + bf16)
  conv_silu_kernel<<<dim3((MM * D_INNER) / 256), blk, 0, stream>>>(xz, conv_w, conv_b, uf, ubf);

  // 3) x_proj: x_dbl = u @ x_proj_w^T  (N=96, zero-padded tile)
  gemm_bf16_wmma<<<dim3(MM / BM, 1), blk, 0, stream>>>(
      ubf, D_INNER, w_xp, D_INNER, xdbl, 96, 96, D_INNER);

  // 3b) bf16 copy of x_dbl for dt GEMM
  cvt(xdbl, xdblbf, (size_t)MM * 96);

  // 4) dt_proj: dtraw = x_dbl[:, :64] @ dt_proj_w^T  (K=64)
  gemm_bf16_wmma<<<dim3(MM / BM, D_INNER / BN), blk, 0, stream>>>(
      xdblbf, 96, w_dt, DT_RANK, dtraw, D_INNER, D_INNER, DT_RANK);

  // 5) selective scan (fuses softplus, skip connection, SiLU gate) -> bf16 y
  scan_kernel<<<dim3(D_INNER / 256, BB), blk, 0, stream>>>(
      uf, dtraw, dt_bias, xdbl, A_log, Dp, xz, ybf);

  // 6) out_proj: out = y @ out_proj_w^T  (N=1024, K=2048)
  gemm_bf16_wmma<<<dim3(MM / BM, D_MODEL / BN), blk, 0, stream>>>(
      ybf, D_INNER, w_out, D_INNER, out, D_MODEL, D_MODEL, D_INNER);
}
